// ResNetAR_80745385165054
// MI455X (gfx1250) — matrix operand order repaired and checked by reading, not compiled, not run on hardware
//
#include <hip/hip_runtime.h>
#include <hip/hip_bf16.h>

typedef __bf16 bf16;
typedef __attribute__((ext_vector_type(16))) __bf16 v16bf;
typedef __attribute__((ext_vector_type(8)))  float  v8f;

#define BM 128
#define BN 64
#define BK 32
#define APAD 8
#define BPAD 8
#define USE_ASYNC_LDS 1

// ---------------------------------------------------------------------------
// Implicit-GEMM conv: out[n,co,ho,wo] = relu( conv(in,w)*scale[co]+bias[co] + res )
// A = im2col(activations)  [M = N*Ho*Wo, K = Cin*KH*KW]   (bf16)
// B = repacked weights     [K, Cout]                       (bf16)
// Block: 256 threads = 8 waves; tile 128x64; wave tile 32x32 = 2x2 WMMA.
// Double-buffered LDS; B tile streamed with global_load_async_to_lds_b128.
// ---------------------------------------------------------------------------
__global__ __launch_bounds__(256)
void conv_wmma(const bf16* __restrict__ in, const bf16* __restrict__ wpk,
               const float* __restrict__ scale, const float* __restrict__ bias,
               const bf16* __restrict__ residual, bf16* __restrict__ out,
               int N, int Cin, int H, int W, int Cout, int Ho, int Wo,
               int KH, int KW, int stride, int pad, int do_relu)
{
    __shared__ bf16 As[2][BM][BK + APAD];
    __shared__ bf16 Bs[2][BK][BN + BPAD];

    const int M    = N * Ho * Wo;
    const int Ktot = Cin * KH * KW;
    const int m0   = blockIdx.x * BM;
    const int n0   = blockIdx.y * BN;
    const int tid  = threadIdx.x;
    const int lane = tid & 31;
    const int wave = tid >> 5;
    const int wm   = (wave & 3) << 5;   // wave row offset (0,32,64,96)
    const int wn   = (wave >> 2) << 5;  // wave col offset (0,32)

    // ---- tile loaders -----------------------------------------------------
    auto loadA = [&](int k0, int buf) {
        for (int idx = tid; idx < BM * BK; idx += 256) {
            int ml = idx >> 5, kl = idx & 31;
            int m = m0 + ml, k = k0 + kl;
            bf16 v = (bf16)0.f;
            if (m < M && k < Ktot) {
                int wo_ = m % Wo; int t = m / Wo;
                int ho_ = t % Ho; int nb = t / Ho;
                int kw_ = k % KW; int t2 = k / KW;
                int kh_ = t2 % KH; int ci = t2 / KH;
                int hh = ho_ * stride - pad + kh_;
                int ww = wo_ * stride - pad + kw_;
                if ((unsigned)hh < (unsigned)H && (unsigned)ww < (unsigned)W)
                    v = in[((long)(nb * Cin + ci) * H + hh) * W + ww];
            }
            As[buf][ml][kl] = v;
        }
    };
    auto loadB = [&](int k0, int buf) {
        // tile = 32 x 64 bf16 = 4096 B = 256 lanes x b128
        if (k0 + BK <= Ktot) {
            int kl = tid >> 3;           // 0..31
            int nl = (tid & 7) * 8;      // 0..56 (8 bf16 per b128)
            const bf16* g = &wpk[(long)(k0 + kl) * Cout + n0 + nl];
#if USE_ASYNC_LDS
            unsigned lofs = (unsigned)(size_t)(void*)&Bs[buf][kl][nl];
            asm volatile("global_load_async_to_lds_b128 %0, %1, off"
                         :: "v"(lofs), "v"(g) : "memory");
#else
            const uint4 d = *(const uint4*)g;
            *(uint4*)&Bs[buf][kl][nl] = d;
#endif
        } else {
            for (int idx = tid; idx < BK * BN; idx += 256) {
                int kl = idx >> 6, nl = idx & 63;
                int k = k0 + kl, n = n0 + nl;
                bf16 v = (bf16)0.f;
                if (k < Ktot && n < Cout) v = wpk[(long)k * Cout + n];
                Bs[buf][kl][nl] = v;
            }
        }
    };

    v8f acc00 = {}, acc01 = {}, acc10 = {}, acc11 = {};

    loadA(0, 0);
    loadB(0, 0);
    int buf = 0;
    for (int k0 = 0; k0 < Ktot; k0 += BK) {
#if USE_ASYNC_LDS
        asm volatile("s_wait_asynccnt 0x0" ::: "memory");
#endif
        __syncthreads();
        if (k0 + BK < Ktot) {           // prefetch next tile into other buffer
            loadA(k0 + BK, buf ^ 1);
            loadB(k0 + BK, buf ^ 1);
        }

        // ---- fragments (contiguous k order, same for A and B) ----
        // A (16-bit 16x32): lanes 0-15 K=0..15       ; lanes 16-31 K=16..31
        // B (16-bit 32x16): lanes 0-15 K=0..15       ; lanes 16-31 K=16..31
        const int half = lane >> 4;
        const int l15  = lane & 15;
        v16bf a0, a1, b0, b1;
        #pragma unroll
        for (int j = 0; j < 16; ++j) {
            int ka = half * 16 + j;
            a0[j] = As[buf][wm + l15][ka];
            a1[j] = As[buf][wm + 16 + l15][ka];
            int kb = half * 16 + j;
            b0[j] = Bs[buf][kb][wn + l15];
            b1[j] = Bs[buf][kb][wn + 16 + l15];
        }
        acc00 = __builtin_amdgcn_wmma_f32_16x16x32_bf16(false, a0, false, b0, (short)0, acc00, false, false);
        acc01 = __builtin_amdgcn_wmma_f32_16x16x32_bf16(false, a0, false, b1, (short)0, acc01, false, false);
        acc10 = __builtin_amdgcn_wmma_f32_16x16x32_bf16(false, a1, false, b0, (short)0, acc10, false, false);
        acc11 = __builtin_amdgcn_wmma_f32_16x16x32_bf16(false, a1, false, b1, (short)0, acc11, false, false);

        __syncthreads();
        buf ^= 1;
    }

    // ---- epilogue: BN scale/bias + optional residual + ReLU ----
    // C layout: VGPR r -> M = r + 8*(lane>>4), N = lane&15
    const int l15  = lane & 15;
    const int mrow = (lane >> 4) * 8;
    v8f* accs[2][2] = {{&acc00, &acc01}, {&acc10, &acc11}};
    #pragma unroll
    for (int ti = 0; ti < 2; ++ti) {
        #pragma unroll
        for (int tj = 0; tj < 2; ++tj) {
            int n = n0 + wn + tj * 16 + l15;
            int mbase = m0 + wm + ti * 16 + mrow;
            if (n < Cout) {
                float sc = scale[n], bi = bias[n];
                v8f& a = *accs[ti][tj];
                #pragma unroll
                for (int r = 0; r < 8; ++r) {
                    int m = mbase + r;
                    if (m < M) {
                        float v = a[r] * sc + bi;
                        int wo_ = m % Wo; int t = m / Wo;
                        int ho_ = t % Ho; int nb = t / Ho;
                        long oidx = ((long)(nb * Cout + n) * Ho + ho_) * Wo + wo_;
                        if (residual) v += (float)residual[oidx];
                        if (do_relu)  v = v > 0.f ? v : 0.f;
                        out[oidx] = (bf16)v;
                    }
                }
            }
        }
    }
}

// ---------------------------------------------------------------------------
// Helpers
// ---------------------------------------------------------------------------
__global__ void fold_bn(const float* g, const float* b, const float* m,
                        const float* v, float* sc, float* bi, int C)
{
    int i = blockIdx.x * blockDim.x + threadIdx.x;
    if (i >= C) return;
    float s = g[i] * rsqrtf(v[i] + 1e-5f);
    sc[i] = s;
    bi[i] = b[i] - m[i] * s;
}

__global__ void repack_w(const float* w, bf16* wpk, int Cout, int Ktot, int rstride)
{
    long i = (long)blockIdx.x * blockDim.x + threadIdx.x;
    if (i >= (long)Cout * Ktot) return;
    int k = (int)(i % Ktot), o = (int)(i / Ktot);
    wpk[(long)k * Cout + o] = (bf16)w[(long)o * rstride + k];
}

// x: (B=2, C=3, S=16, H, W) f32  ->  (B*S, C, H, W) bf16  (bakes the transpose)
__global__ void stem_pack(const float* x, bf16* out, int B, int C, int S, int HW)
{
    long i = (long)blockIdx.x * blockDim.x + threadIdx.x;
    long total = (long)B * C * S * HW;
    if (i >= total) return;
    long p = i % HW; long t = i / HW;
    int c = (int)(t % C); t /= C;
    int s = (int)(t % S); int b = (int)(t / S);
    out[i] = (bf16)x[(((long)b * C + c) * S + s) * HW + p];
}

__global__ void maxpool_k(const bf16* in, bf16* out, int NC, int H, int W, int Ho, int Wo)
{
    long i = (long)blockIdx.x * blockDim.x + threadIdx.x;
    long total = (long)NC * Ho * Wo;
    if (i >= total) return;
    int wo = (int)(i % Wo); long t = i / Wo;
    int ho = (int)(t % Ho); int nc = (int)(t / Ho);
    float mx = -3.4e38f;
    for (int kh = 0; kh < 3; ++kh)
        for (int kw = 0; kw < 3; ++kw) {
            int h = ho * 2 - 1 + kh, w = wo * 2 - 1 + kw;
            if ((unsigned)h < (unsigned)H && (unsigned)w < (unsigned)W) {
                float v = (float)in[((long)nc * H + h) * W + w];
                mx = v > mx ? v : mx;
            }
        }
    out[i] = (bf16)mx;
}

__global__ void fill_k(float* p, float v, int n)
{
    int i = blockIdx.x * blockDim.x + threadIdx.x;
    if (i < n) p[i] = v;
}

// yn: normalize y over C per (b,s,h,w); store transposed [b][p][s][c] for fast dots
__global__ void norm_k(const bf16* y, float* ynT, int BS, int C, int P, int S)
{
    int i = blockIdx.x * blockDim.x + threadIdx.x;
    if (i >= BS * P) return;
    int p = i % P, bs = i / P;
    float ss = 0.f;
    for (int c = 0; c < C; ++c) {
        float v = (float)y[((long)bs * C + c) * P + p];
        ss += v * v;
    }
    float inv = 1.f / (sqrtf(ss) + 1e-6f);
    int s = bs % S, b = bs / S;
    long base = (((long)b * P + p) * S + s) * (long)C;
    for (int c = 0; c < C; ++c)
        ynT[base + c] = (float)y[((long)bs * C + c) * P + p] * inv;
}

// A[b,p,i,j] = relu( yn[b,i,:,p] . yn[b,j,:,p] )
__global__ void attn_k(const float* ynT, float* A, int B, int P, int S, int C)
{
    int i = blockIdx.x * blockDim.x + threadIdx.x;
    if (i >= B * P * S * S) return;
    int j  = i % S; int t = i / S;
    int ii = t % S; t /= S;
    int p  = t % P; int b = t / P;
    const float* r1 = ynT + (((long)b * P + p) * S + ii) * C;
    const float* r2 = ynT + (((long)b * P + p) * S + j)  * C;
    float d = 0.f;
    for (int c = 0; c < C; ++c) d += r1[c] * r2[c];
    A[i] = d > 0.f ? d : 0.f;
}

// causal softmax over i<s and context mix:  cs[b,c,p] = sum_i w_i * buf[i,b,c,p]
__global__ void cs_k(const float* A, const float* buf, float* cs,
                     int s, int B, int C, int P, int S)
{
    int i = blockIdx.x * blockDim.x + threadIdx.x;
    if (i >= B * C * P) return;
    int p = i % P; int t = i / P;
    int c = t % C; int b = t / C;
    float mx = -3.4e38f;
    for (int ii = 0; ii < s; ++ii) {
        float l = A[(((long)b * P + p) * S + ii) * S + s];
        mx = l > mx ? l : mx;
    }
    if (s == 0) mx = 0.f;
    float sum = 0.f, acc = 0.f;
    for (int ii = 0; ii < s; ++ii) {
        float e = expf(A[(((long)b * P + p) * S + ii) * S + s] - mx);
        sum += e;
        acc += e * buf[(((long)ii * B + b) * C + c) * P + p];
    }
    float w = 1.f / (sum > 1e-30f ? sum : 1e-30f);
    cs[((long)b * C + c) * P + p] = acc * w;
}

// yo = W1y[b,s] + W2 @ cs ; write into buf[s]
__global__ void yo_k(const bf16* w1y, const float* ce, const float* cs,
                     float* buf, int s, int B, int C, int P, int S, int CE)
{
    int i = blockIdx.x * blockDim.x + threadIdx.x;
    if (i >= B * C * P) return;
    int p = i % P; int t = i / P;
    int o = t % C; int b = t / C;
    float acc = (float)w1y[(((long)(b * S + s)) * C + o) * P + p];
    const float* w2 = ce + (long)o * CE + C;   // second half of conv_end row
    for (int c = 0; c < C; ++c)
        acc += w2[c] * cs[((long)b * C + c) * P + p];
    buf[(((long)s * B + b) * C + o) * P + p] = acc;
}

// bn_end + relu + spatial mean -> h0[(b*S+s)*C + c]
__global__ void head_k(const float* buf, const float* sc, const float* bi,
                       float* h0, int B, int S, int C, int P)
{
    int i = blockIdx.x * blockDim.x + threadIdx.x;
    if (i >= B * S * C) return;
    int c = i % C; int t = i / C;
    int s = t % S; int b = t / S;
    float a = 0.f;
    for (int p = 0; p < P; ++p) {
        float v = buf[(((long)s * B + b) * C + c) * P + p] * sc[c] + bi[c];
        a += v > 0.f ? v : 0.f;
    }
    h0[i] = a / (float)P;
}

__global__ void fc_k(const float* h0, const float* w, const float* bvec,
                     float* h, int R, int C)
{
    int i = blockIdx.x * blockDim.x + threadIdx.x;
    if (i >= R * C) return;
    int o = i % C, r = i / C;
    float acc = bvec[o];
    for (int c = 0; c < C; ++c) acc += h0[(long)r * C + c] * w[(long)o * C + c];
    h[i] = acc > 0.f ? acc : 0.f;
}

__global__ void out_k(const float* h, const float* fow, const float* fob,
                      const float* faw, const float* fab, float* out, int R, int C)
{
    int r = blockIdx.x * blockDim.x + threadIdx.x;
    if (r >= R) return;
    float a = fob[0], g = fab[0];
    for (int c = 0; c < C; ++c) {
        a += h[(long)r * C + c] * fow[c];
        g += h[(long)r * C + c] * faw[c];
    }
    out[r]     = a;   // offset
    out[R + r] = g;   // angle
}

// ---------------------------------------------------------------------------
static inline int cdiv(long a, long b) { return (int)((a + b - 1) / b); }

extern "C" void kernel_launch(void* const* d_in, const int* in_sizes, int n_in,
                              void* d_out, int out_size, void* d_ws, size_t ws_size,
                              hipStream_t stream)
{
    (void)in_sizes; (void)n_in; (void)out_size; (void)ws_size;
    const int NB = 32;   // B*S frames
    int ip = 0;
    auto F = [&]() -> const float* { return (const float*)d_in[ip++]; };

    // ---- workspace bump allocator ----
    char* wsp = (char*)d_ws;
    size_t off = 0;
    auto alloc = [&](size_t bytes) -> void* {
        off = (off + 255) & ~(size_t)255;
        void* p = wsp + off;
        off += bytes;
        return p;
    };
    bf16*  xbf  = (bf16*)alloc((size_t)NB * 3 * 224 * 224 * 2);
    bf16*  big1 = (bf16*)alloc((size_t)NB * 64 * 112 * 112 * 2);
    bf16*  big2 = (bf16*)alloc((size_t)NB * 64 * 112 * 112 * 2);
    bf16*  midb = (bf16*)alloc((size_t)NB * 64 * 56 * 56 * 2);
    bf16*  tmpb = (bf16*)alloc((size_t)NB * 128 * 28 * 28 * 2);
    bf16*  wpk  = (bf16*)alloc((size_t)512 * 4608 * 2);
    float* scb  = (float*)alloc(512 * 4);
    float* bib  = (float*)alloc(512 * 4);
    float* ones = (float*)alloc(512 * 4);
    float* zers = (float*)alloc(512 * 4);
    bf16*  w1y  = (bf16*)alloc((size_t)NB * 512 * 49 * 2);
    float* ynT  = (float*)alloc((size_t)2 * 49 * 16 * 512 * 4);
    float* Aat  = (float*)alloc((size_t)2 * 49 * 16 * 16 * 4);
    float* bufS = (float*)alloc((size_t)16 * 2 * 512 * 49 * 4);
    float* csb  = (float*)alloc((size_t)2 * 512 * 49 * 4);
    float* h0   = (float*)alloc((size_t)NB * 512 * 4);
    float* hb   = (float*)alloc((size_t)NB * 512 * 4);

    auto conv = [&](const bf16* in, const float* w,
                    const float* g, const float* b, const float* m, const float* v,
                    bf16* out, const bf16* res,
                    int N, int Cin, int H, int W, int Cout,
                    int KH, int KW, int st, int pad, int relu) {
        int Ho = (H + 2 * pad - KH) / st + 1;
        int Wo = (W + 2 * pad - KW) / st + 1;
        int Ktot = Cin * KH * KW;
        repack_w<<<cdiv((long)Cout * Ktot, 256), 256, 0, stream>>>(w, wpk, Cout, Ktot, Ktot);
        fold_bn<<<cdiv(Cout, 256), 256, 0, stream>>>(g, b, m, v, scb, bib, Cout);
        int M = N * Ho * Wo;
        dim3 grid(cdiv(M, BM), cdiv(Cout, BN));
        conv_wmma<<<grid, 256, 0, stream>>>(in, wpk, scb, bib, res, out,
                                            N, Cin, H, W, Cout, Ho, Wo, KH, KW, st, pad, relu);
    };

    // ---- stem ----
    const float* Xp  = F();
    const float* c1w = F();
    const float* g1 = F(); const float* b1 = F(); const float* m1 = F(); const float* v1 = F();
    stem_pack<<<cdiv((long)NB * 3 * 224 * 224, 256), 256, 0, stream>>>(Xp, xbf, 2, 3, 16, 224 * 224);
    conv(xbf, c1w, g1, b1, m1, v1, big1, nullptr, NB, 3, 224, 224, 64, 7, 7, 2, 3, 1);
    maxpool_k<<<cdiv((long)NB * 64 * 56 * 56, 256), 256, 0, stream>>>(big1, big2, NB * 64, 112, 112, 56, 56);

    // ---- residual blocks ----
    const int planes[8]  = {64, 64, 128, 128, 256, 256, 512, 512};
    const int strides[8] = {1, 1, 2, 1, 2, 1, 2, 1};
    bf16* cur = big2; bf16* oth = big1;
    int curC = 64, curH = 56;
    for (int bi = 0; bi < 8; ++bi) {
        int pl = planes[bi], st = strides[bi];
        int Hout = (st == 2) ? curH / 2 : curH;
        const float* w1 = F();
        const float* gg1 = F(); const float* bb1 = F(); const float* mm1 = F(); const float* vv1 = F();
        const float* w2 = F();
        const float* gg2 = F(); const float* bb2 = F(); const float* mm2 = F(); const float* vv2 = F();
        bool down = (st != 1) || (curC != pl);
        const bf16* res;
        if (down) {
            const float* dw = F();
            const float* dg = F(); const float* db = F(); const float* dm = F(); const float* dv = F();
            conv(cur, dw, dg, db, dm, dv, tmpb, nullptr, NB, curC, curH, curH, pl, 1, 1, st, 0, 0);
            res = tmpb;
        } else {
            res = cur;
        }
        conv(cur,  w1, gg1, bb1, mm1, vv1, midb, nullptr, NB, curC, curH, curH, pl, 3, 3, st, 1, 1);
        conv(midb, w2, gg2, bb2, mm2, vv2, oth,  res,     NB, pl,   Hout, Hout, pl, 3, 3, 1,  1, 1);
        bf16* t = cur; cur = oth; oth = t;
        curC = pl; curH = Hout;
    }
    // cur = y : (32, 512, 7, 7) bf16

    // ---- tail params ----
    const float* ce  = F();                                         // conv_end (512,1024)
    const float* ge = F(); const float* be = F(); const float* me = F(); const float* ve = F();
    const float* fcw = F(); const float* fcb = F();
    const float* fow = F(); const float* fob = F();
    const float* faw = F(); const float* fab = F();

    // W1 @ y for all frames as 1x1 conv through the WMMA kernel
    fill_k<<<2, 256, 0, stream>>>(ones, 1.f, 512);
    fill_k<<<2, 256, 0, stream>>>(zers, 0.f, 512);
    repack_w<<<cdiv((long)512 * 512, 256), 256, 0, stream>>>(ce, wpk, 512, 512, 1024);
    {
        int M = NB * 49;
        dim3 grid(cdiv(M, BM), cdiv(512, BN));
        conv_wmma<<<grid, 256, 0, stream>>>(cur, wpk, ones, zers, nullptr, w1y,
                                            NB, 512, 7, 7, 512, 7, 7, 1, 1, 1, 0, 0);
    }

    // attention matrix
    norm_k<<<cdiv(NB * 49, 256), 256, 0, stream>>>(cur, ynT, NB, 512, 49, 16);
    attn_k<<<cdiv((long)2 * 49 * 16 * 16, 256), 256, 0, stream>>>(ynT, Aat, 2, 49, 16, 512);

    // causal scan over S
    for (int s = 0; s < 16; ++s) {
        cs_k<<<cdiv((long)2 * 512 * 49, 256), 256, 0, stream>>>(Aat, bufS, csb, s, 2, 512, 49, 16);
        yo_k<<<cdiv((long)2 * 512 * 49, 256), 256, 0, stream>>>(w1y, ce, csb, bufS, s, 2, 512, 49, 16, 1024);
    }

    // bn_end + relu + mean, FCs, heads
    fold_bn<<<cdiv(512, 256), 256, 0, stream>>>(ge, be, me, ve, scb, bib, 512);
    head_k<<<cdiv((long)NB * 512, 256), 256, 0, stream>>>(bufS, scb, bib, h0, 2, 16, 512, 49);
    fc_k<<<cdiv((long)NB * 512, 256), 256, 0, stream>>>(h0, fcw, fcb, hb, NB, 512);
    out_k<<<1, 32, 0, stream>>>(hb, fow, fob, faw, fab, (float*)d_out, NB, 512);
}
